// MultiHeadAttention_36421322670164
// MI455X (gfx1250) — matrix operand order repaired and checked by reading, not compiled, not run on hardware
//
#include <hip/hip_runtime.h>
#include <hip/hip_bf16.h>
#include <math.h>

typedef __attribute__((ext_vector_type(16))) _Float16 v16h;
typedef __attribute__((ext_vector_type(8)))  _Float16 v8h;
typedef __attribute__((ext_vector_type(8)))  float    v8f;

__device__ __forceinline__ v8f wmma_f16(v16h a, v16h b, v8f c) {
    // (neg_a, A, neg_b, B, c_mod, C, reuse_a, reuse_b)
    return __builtin_amdgcn_wmma_f32_16x16x32_f16(false, a, false, b, (short)0, c, false, false);
}

// ---------------- elementwise converts ----------------

__global__ void cvt_f32_to_f16(const float* __restrict__ x, _Float16* __restrict__ y, int n) {
    int i = blockIdx.x * blockDim.x + threadIdx.x;
    if (i < n) y[i] = (_Float16)x[i];
}

// Wt[n*C + k] = W[k*C + n]  (store B^T so GEMM B-tiles are contiguous)
__global__ void cvt_transpose_f32_to_f16(const float* __restrict__ W, _Float16* __restrict__ Wt, int C) {
    int i = blockIdx.x * blockDim.x + threadIdx.x;
    if (i < C * C) {
        int k = i / C, n = i % C;
        Wt[(size_t)n * C + k] = (_Float16)W[i];
    }
}

// ---------------- 64x64-per-wave register-blocked WMMA GEMM ----------------
// A: [M,K] f16 row-major.  Bt: [N,K] f16 row-major (i.e. B transposed).
// Each wave computes a 64x64 output block: 4x4 grid of 16x16 WMMA tiles.
// 16 WMMAs per 32-deep K step vs only 8 v16h operand loads (4x A, 4x B).
// __launch_bounds__(128,1): let the allocator use the full 256-VGPR budget
// (16 v8f accumulators = 128 VGPRs live) -- avoids scratch spills.
// out_mode 0: Oh[M,N] f16 row-major
// out_mode 1: Oh[N,M] f16 (transposed store, for V)
// out_mode 2: Of[M,N] f32 + bias[n]
__global__ void __launch_bounds__(128, 1)
gemm64_wmma(const _Float16* __restrict__ A, const _Float16* __restrict__ Bt,
            _Float16* __restrict__ Oh, float* __restrict__ Of,
            const float* __restrict__ bias,
            int M, int N, int K, int out_mode) {
    const int tid  = threadIdx.x;
    const int lane = tid & 31;
    const int wave = tid >> 5;               // 4 waves / block
    const int l15  = lane & 15;
    const int grp  = lane >> 4;
    const int col0 = blockIdx.x * 64;
    const int row0 = blockIdx.y * 256 + wave * 64;

    // A layout: lane = M row; lanes 0-15 hold K=0..15, lanes 16-31 hold K=16..31 (contiguous)
    const _Float16* pa = A  + (size_t)(row0 + l15) * K + grp * 16;
    // B layout: lane = N col; lanes 0-15 hold K=0..15, lanes 16-31 hold K=16..31 (contiguous in Bt row)
    const _Float16* pb = Bt + (size_t)(col0 + l15) * K + grp * 16;
    const size_t rstride = (size_t)16 * K;   // 16 rows/cols between sub-tiles

    v8f acc[4][4] = {};
#pragma unroll 1
    for (int k0 = 0; k0 < K; k0 += 32) {
        // A panel: 4 row sub-tiles, loaded once and reused across 4 B tiles each
        v16h a[4];
#pragma unroll
        for (int i = 0; i < 4; ++i) {
            const _Float16* p = pa + (size_t)i * rstride + k0;
#pragma unroll
            for (int j = 0; j < 16; ++j) { a[i][j] = p[j]; }
        }
        __builtin_prefetch(pa + k0 + 128, 0, 1);
        __builtin_prefetch(pb + k0 + 128, 0, 1);
        // stream B column tiles: one v16h live at a time -> 4 WMMAs each
#pragma unroll
        for (int j = 0; j < 4; ++j) {
            v16h b = *(const v16h*)(pb + (size_t)j * rstride + k0);
            acc[0][j] = wmma_f16(a[0], b, acc[0][j]);
            acc[1][j] = wmma_f16(a[1], b, acc[1][j]);
            acc[2][j] = wmma_f16(a[2], b, acc[2][j]);
            acc[3][j] = wmma_f16(a[3], b, acc[3][j]);
        }
    }

    const int mb = grp * 8;                  // C/D layout: VGPR r -> M = mb + r (within sub-tile)
    if (out_mode == 2) {
#pragma unroll
        for (int j = 0; j < 4; ++j) {
            const int n = col0 + j * 16 + l15;
            const float bb = bias[n];
#pragma unroll
            for (int i = 0; i < 4; ++i)
#pragma unroll
                for (int r = 0; r < 8; ++r)
                    Of[(size_t)(row0 + i * 16 + mb + r) * N + n] = acc[i][j][r] + bb;
        }
    } else if (out_mode == 1) {
#pragma unroll
        for (int j = 0; j < 4; ++j) {
            const int n = col0 + j * 16 + l15;
#pragma unroll
            for (int i = 0; i < 4; ++i)
#pragma unroll
                for (int r = 0; r < 8; ++r)
                    Oh[(size_t)n * M + (row0 + i * 16 + mb + r)] = (_Float16)acc[i][j][r];
        }
    } else {
#pragma unroll
        for (int j = 0; j < 4; ++j) {
            const int n = col0 + j * 16 + l15;
#pragma unroll
            for (int i = 0; i < 4; ++i)
#pragma unroll
                for (int r = 0; r < 8; ++r)
                    Oh[(size_t)(row0 + i * 16 + mb + r) * N + n] = (_Float16)acc[i][j][r];
        }
    }
}

// ---------------- fused masked-softmax attention ----------------
// One wave handles 16 query rows of one (b,h). LDS holds the 16 x T logit strip.
__global__ void __launch_bounds__(32, 1)
attn_fused(const _Float16* __restrict__ Qp, const _Float16* __restrict__ Kp,
           const _Float16* __restrict__ VpT, const int* __restrict__ mask,
           _Float16* __restrict__ AO,
           int Bn, int T, int Cn, int H) {
    extern __shared__ _Float16 S[];   // [16][T] unnormalized probs
    __shared__ float rowscale[16];

    const int lane = threadIdx.x;
    const int l15  = lane & 15;
    const int grp  = lane >> 4;
    const int TB   = T / 16;
    const int blk  = blockIdx.x;
    const int qt   = blk % TB;
    const int hb   = blk / TB;
    const int h    = hb % H;
    const int b    = hb / H;
    const int Dh   = Cn / H;                  // 64
    const size_t Mtot = (size_t)Bn * T;

    const _Float16* Qh = Qp + (size_t)b * T * Cn + (size_t)h * Dh;
    const _Float16* Kh = Kp + (size_t)b * T * Cn + (size_t)h * Dh;
    const int* mrow = mask + (size_t)b * T * T + (size_t)(qt * 16) * T;

    // Q A-tiles: 16 rows x 64 features = two 16x32 slices, held in registers
    v16h aq0, aq1;
    {
        const _Float16* pa = Qh + (size_t)(qt * 16 + l15) * Cn + grp * 16;
#pragma unroll
        for (int j = 0; j < 16; ++j) {
            aq0[j] = pa[j];
            aq1[j] = pa[32 + j];
        }
    }
    const float scale = rsqrtf((float)Cn);    // C**-0.5 per reference

    // ---- Phase 1: S = mask(scale * Q K^T) -> LDS ----
    for (int kt = 0; kt < TB; ++kt) {
        // B-tile of K^T: column = key index, K-dim = feature (contiguous per key row)
        const _Float16* pk = Kh + (size_t)(kt * 16 + l15) * Cn + grp * 16;
        v16h b0 = *(const v16h*)(pk);
        v16h b1 = *(const v16h*)(pk + 32);
        v8f acc = {};
        acc = wmma_f16(aq0, b0, acc);
        acc = wmma_f16(aq1, b1, acc);
        const int ncol  = kt * 16 + l15;
        const int mbase = grp * 8;
#pragma unroll
        for (int r = 0; r < 8; ++r) {
            const int m = mbase + r;
            float sv = acc[r] * scale;
            if (mrow[(size_t)m * T + ncol] == 0) sv = -30000.0f;
            S[m * T + ncol] = (_Float16)sv;
        }
    }

    // ---- Phase 2: row softmax (unnormalized exp in LDS, 1/sum in rowscale) ----
    for (int r = 0; r < 16; ++r) {
        float mmax = -1.0e30f;
        for (int c = lane; c < T; c += 32) mmax = fmaxf(mmax, (float)S[r * T + c]);
#pragma unroll
        for (int off = 16; off > 0; off >>= 1) mmax = fmaxf(mmax, __shfl_xor(mmax, off, 32));
        float sum = 0.0f;
        for (int c = lane; c < T; c += 32) {
            float e = __expf((float)S[r * T + c] - mmax);
            sum += e;
            S[r * T + c] = (_Float16)e;
        }
#pragma unroll
        for (int off = 16; off > 0; off >>= 1) sum += __shfl_xor(sum, off, 32);
        if (lane == 0) rowscale[r] = 1.0f / sum;
    }

    // ---- Phase 3: O = P @ V_head (Dh = 64 -> 4 accumulators) ----
    v8f acc0 = {}, acc1 = {}, acc2 = {}, acc3 = {};
    const _Float16* pvbase = VpT + (size_t)(h * Dh + l15) * Mtot + (size_t)b * T;
#pragma unroll 1
    for (int kt = 0; kt < T / 32; ++kt) {
        v16h ap;
        const _Float16* pr = S + l15 * T + kt * 32 + grp * 16;
#pragma unroll
        for (int j = 0; j < 16; ++j) { ap[j] = pr[j]; }
        const size_t ko = (size_t)(kt * 32 + grp * 16);
        acc0 = wmma_f16(ap, *(const v16h*)(pvbase + (size_t)0  * Mtot + ko), acc0);
        acc1 = wmma_f16(ap, *(const v16h*)(pvbase + (size_t)16 * Mtot + ko), acc1);
        acc2 = wmma_f16(ap, *(const v16h*)(pvbase + (size_t)32 * Mtot + ko), acc2);
        acc3 = wmma_f16(ap, *(const v16h*)(pvbase + (size_t)48 * Mtot + ko), acc3);
    }

    _Float16* Ohp = AO + (size_t)b * T * Cn + (size_t)h * Dh;
    const int mbase = grp * 8;
#pragma unroll
    for (int nt = 0; nt < 4; ++nt) {
        v8f a = (nt == 0) ? acc0 : (nt == 1) ? acc1 : (nt == 2) ? acc2 : acc3;
#pragma unroll
        for (int r = 0; r < 8; ++r) {
            const int m = mbase + r;
            float o = a[r] * rowscale[m];
            Ohp[(size_t)(qt * 16 + m) * Cn + nt * 16 + l15] = (_Float16)o;
        }
    }
}

// ---------------- launch ----------------

extern "C" void kernel_launch(void* const* d_in, const int* in_sizes, int n_in,
                              void* d_out, int out_size, void* d_ws, size_t ws_size,
                              hipStream_t stream) {
    (void)in_sizes; (void)n_in; (void)out_size; (void)ws_size;
    const int B = 2, T = 2048, C = 1024, H = 16;
    const int M = B * T;                       // 4096

    const float* query = (const float*)d_in[0];
    const float* key   = (const float*)d_in[1];
    const float* value = (const float*)d_in[2];
    const int*   mask  = (const int*)d_in[3];
    const float* Wq    = (const float*)d_in[4];
    const float* Wk    = (const float*)d_in[5];
    const float* Wv    = (const float*)d_in[6];
    const float* Wp    = (const float*)d_in[7];
    const float* bp    = (const float*)d_in[8];
    float* out = (float*)d_out;

    _Float16* w = (_Float16*)d_ws;
    const size_t NTC = (size_t)M * C;          // 4M halfs each
    const size_t WSZ = (size_t)C * C;          // 1M halfs each
    _Float16* q16 = w; w += NTC;
    _Float16* k16 = w; w += NTC;
    _Float16* v16 = w; w += NTC;
    _Float16* wqt = w; w += WSZ;
    _Float16* wkt = w; w += WSZ;
    _Float16* wvt = w; w += WSZ;
    _Float16* wpt = w; w += WSZ;
    _Float16* Qp  = w; w += NTC;
    _Float16* Kp  = w; w += NTC;
    _Float16* VpT = w; w += NTC;               // [C, M] transposed V projection
    _Float16* AO  = w; w += NTC;

    // f32 -> f16 activation converts
    cvt_f32_to_f16<<<(int)(NTC / 256), 256, 0, stream>>>(query, q16, (int)NTC);
    cvt_f32_to_f16<<<(int)(NTC / 256), 256, 0, stream>>>(key,   k16, (int)NTC);
    cvt_f32_to_f16<<<(int)(NTC / 256), 256, 0, stream>>>(value, v16, (int)NTC);
    // weight converts (transposed for contiguous WMMA B-tile loads)
    cvt_transpose_f32_to_f16<<<(int)(WSZ / 256), 256, 0, stream>>>(Wq, wqt, C);
    cvt_transpose_f32_to_f16<<<(int)(WSZ / 256), 256, 0, stream>>>(Wk, wkt, C);
    cvt_transpose_f32_to_f16<<<(int)(WSZ / 256), 256, 0, stream>>>(Wv, wvt, C);
    cvt_transpose_f32_to_f16<<<(int)(WSZ / 256), 256, 0, stream>>>(Wp, wpt, C);

    // projections: Q, K row-major f16; V stored transposed [C, M]
    // grid: 64-wide col blocks, 256-tall row blocks (4 waves x 64 rows)
    dim3 g1(C / 64, M / 256);
    gemm64_wmma<<<g1, 128, 0, stream>>>(q16, wqt, Qp,  nullptr, nullptr, M, C, C, 0);
    gemm64_wmma<<<g1, 128, 0, stream>>>(k16, wkt, Kp,  nullptr, nullptr, M, C, C, 0);
    gemm64_wmma<<<g1, 128, 0, stream>>>(v16, wvt, VpT, nullptr, nullptr, M, C, C, 1);

    // fused attention: one wave per 16-row query block per head
    attn_fused<<<B * H * (T / 16), 32, 16 * T * (int)sizeof(_Float16), stream>>>(
        Qp, Kp, VpT, mask, AO, B, T, C, H);

    // output projection + bias -> f32
    gemm64_wmma<<<g1, 128, 0, stream>>>(AO, wpt, nullptr, out, bp, M, C, C, 2);
}
